// Head_68229850464827
// MI455X (gfx1250) — matrix operand order repaired and checked by reading, not compiled, not run on hardware
//
#include <hip/hip_runtime.h>
#include <hip/hip_bf16.h>
#include <math.h>

#define BB 4
#define TT 2048
#define CC 1024
#define HH 64

typedef unsigned short u16;
typedef __attribute__((ext_vector_type(16))) __bf16 v16bf;
typedef __attribute__((ext_vector_type(8)))  float  v8f;

union BfVec { v16bf v; uint4 q[2]; u16 s[16]; };

#if defined(__has_builtin)
#if __has_builtin(__builtin_amdgcn_cvt_pk_bf16_f32)
#define HAVE_CVT_PK_BF16 1
#endif
#endif

// pack two f32 into packed bf16 {lo=a, hi=b}
__device__ inline unsigned int pack2(float a, float b) {
#ifdef HAVE_CVT_PK_BF16
    auto r = __builtin_amdgcn_cvt_pk_bf16_f32(a, b);   // v_cvt_pk_bf16_f32
    unsigned int u;
    __builtin_memcpy(&u, &r, 4);
    return u;
#elif defined(__gfx1250__)
    // CDNA5 VOP3P opcodes 62/63: single-instruction f32->bf16 converts
    unsigned int d;
    asm("v_fma_mixlo_bf16 %0, %1, 1.0, 0" : "=v"(d) : "v"(a));
    asm("v_fma_mixhi_bf16 %0, %1, 1.0, 0" : "+v"(d) : "v"(b));
    return d;
#else
    unsigned int ua = __float_as_uint(a);
    ua += 0x7FFFu + ((ua >> 16) & 1u);
    unsigned int ub = __float_as_uint(b);
    ub += 0x7FFFu + ((ub >> 16) & 1u);
    return (ua >> 16) | (ub & 0xFFFF0000u);
#endif
}
__device__ inline u16 f2bf(float f) {
#if defined(HAVE_CVT_PK_BF16)
    return (u16)pack2(f, 0.0f);
#elif defined(__gfx1250__)
    unsigned int d;
    asm("v_fma_mixlo_bf16 %0, %1, 1.0, 0" : "=v"(d) : "v"(f));
    return (u16)d;
#else
    unsigned int u = __float_as_uint(f);
    u += 0x7FFFu + ((u >> 16) & 1u);
    return (u16)(u >> 16);
#endif
}
// B operand (32x16 bf16): per lane a contiguous 16-element (32B) K-run
__device__ inline v16bf load_b_op(const u16* p) {
    BfVec t;
    t.q[0] = *(const uint4*)(p);
    t.q[1] = *(const uint4*)(p + 8);
    return t.v;
}
// A operand (16x32 bf16): per lane two 8-element (16B) K-runs
__device__ inline v16bf load_a_op(const u16* p0, const u16* p1) {
    BfVec t;
    t.q[0] = *(const uint4*)p0;
    t.q[1] = *(const uint4*)p1;
    return t.v;
}

// ---------------- Phase 0: weights -> bf16, transposed [H][C] ----------------
__global__ void wcvt_kernel(const float* __restrict__ Wk, const float* __restrict__ Wq,
                            const float* __restrict__ Wv,
                            u16* __restrict__ Wqt, u16* __restrict__ Wkt, u16* __restrict__ Wvt) {
    int idx = blockIdx.x * blockDim.x + threadIdx.x;     // over H*C
    if (idx >= HH * CC) return;
    int h = idx / CC, c = idx - h * CC;
    Wqt[h * CC + c] = f2bf(Wq[c * HH + h]);
    Wkt[h * CC + c] = f2bf(Wk[c * HH + h]);
    Wvt[h * CC + c] = f2bf(Wv[c * HH + h]);
}

// ---------------- Phase 1: QKV projection (bf16 WMMA, f32 accum) -------------
// one wave = one (batch, 16-row, 16-h) tile of Q, K, V simultaneously
__global__ void __launch_bounds__(256) qkv_proj_kernel(
    const float* __restrict__ x,
    const u16* __restrict__ Wqt, const u16* __restrict__ Wkt, const u16* __restrict__ Wvt,
    u16* __restrict__ Qbf, u16* __restrict__ Kbf, u16* __restrict__ Vt) {
    const int lane = threadIdx.x & 31;
    const int wid  = threadIdx.x >> 5;
    const int tile = blockIdx.x * 8 + wid;   // BB * 128 * 4 = 2048 tiles
    const int ht = tile & 3;
    const int rt = (tile >> 2) & 127;
    const int b  = tile >> 9;
    const int h0 = ht * 16;
    const int m0 = rt * 16;
    const int n  = lane & 15;
    const int hi = lane >> 4;

    v8f cq = {}; v8f ck = {}; v8f cv = {};

    const float* xrow = x + (size_t)(b * TT + m0 + n) * CC;   // A row = lane&15
    for (int c0 = 0; c0 < CC; c0 += 32) {
        // A operand from x (f32 -> bf16), K runs {16*hi, 16*hi+8}
        const float* p0 = xrow + c0 + hi * 16;
        const float* p1 = xrow + c0 + hi * 16 + 8;
        float4 f0 = ((const float4*)p0)[0];
        float4 f1 = ((const float4*)p0)[1];
        float4 f2 = ((const float4*)p1)[0];
        float4 f3 = ((const float4*)p1)[1];
        BfVec a;
        a.q[0] = make_uint4(pack2(f0.x,f0.y), pack2(f0.z,f0.w), pack2(f1.x,f1.y), pack2(f1.z,f1.w));
        a.q[1] = make_uint4(pack2(f2.x,f2.y), pack2(f2.z,f2.w), pack2(f3.x,f3.y), pack2(f3.z,f3.w));

        const int wb = (h0 + n) * CC + c0 + hi * 16;          // B col = lane&15, contiguous K
        v16bf bq = load_b_op(Wqt + wb);
        v16bf bk = load_b_op(Wkt + wb);
        v16bf bv = load_b_op(Wvt + wb);
        cq = __builtin_amdgcn_wmma_f32_16x16x32_bf16(false, a.v, false, bq, (short)0, cq, false, false);
        ck = __builtin_amdgcn_wmma_f32_16x16x32_bf16(false, a.v, false, bk, (short)0, ck, false, false);
        cv = __builtin_amdgcn_wmma_f32_16x16x32_bf16(false, a.v, false, bv, (short)0, cv, false, false);
    }
    // C/D layout: M = v + 8*hi, N = lane&15
    for (int v = 0; v < 8; ++v) {
        int row = m0 + hi * 8 + v;
        size_t qk = (size_t)(b * TT + row) * HH + h0 + n;
        Qbf[qk] = f2bf(cq[v]);
        Kbf[qk] = f2bf(ck[v]);
        Vt[(size_t)b * HH * TT + (size_t)(h0 + n) * TT + row] = f2bf(cv[v]);  // transposed
    }
}

// ---------------- Phase 2: causal flash attention ----------------------------
// paired waves (phase 0/1) split the key stream of one 16-row query tile:
// phase p handles key blocks j = 32*p, 32*p+64, ... ; merged via LDS at the end.
__global__ void __launch_bounds__(256) attn_kernel(
    const u16* __restrict__ Qbf, const u16* __restrict__ Kbf,
    const u16* __restrict__ Vt, float* __restrict__ out) {
    __shared__ u16   sP[8 * 16 * 32];       // per-wave P transpose tile (8KB)
    __shared__ float sO[4][16][64];         // per-pair partial O (16KB)
    __shared__ float sML[4][2][16];         // per-pair row max / row sum

    const int lane  = threadIdx.x & 31;
    const int wid   = threadIdx.x >> 5;
    const int pair  = wid >> 1;             // 4 tiles per block
    const int phase = wid & 1;
    const int tile  = blockIdx.x * 4 + pair;  // BB * 128 = 512 tiles
    const int rt = tile & 127;
    const int b  = tile >> 7;
    const int m0 = rt * 16;
    const int n  = lane & 15;
    const int hi = lane >> 4;
    u16* myP = sP + wid * (16 * 32);

    const u16* Qb = Qbf + (size_t)b * TT * HH;
    const u16* Kb = Kbf + (size_t)b * TT * HH;
    const u16* Vb = Vt  + (size_t)b * HH * TT;

    // Q tile as two A-operands (h chunks 0..31 and 32..63), resident all kernel
    const u16* qrow = Qb + (size_t)(m0 + n) * HH;
    v16bf aq0 = load_a_op(qrow + hi * 16,      qrow + 8 + hi * 16);
    v16bf aq1 = load_a_op(qrow + 32 + hi * 16, qrow + 40 + hi * 16);

    v8f o0 = {}, o1 = {}, o2 = {}, o3 = {};
    float mi[8], li[8];
    for (int v = 0; v < 8; ++v) { mi[v] = -INFINITY; li[v] = 0.0f; }

    // logits pre-scaled to base-2: C^-0.5 * log2(e)
    const float kscale = 0.03125f * 1.44269504088896340736f;

    for (int j = 32 * phase; j <= m0 + 15; j += 64) {
        // prefetch this wave's next key block (gfx1250 global_prefetch_b8)
        if (j + 64 <= m0 + 15)
            __builtin_prefetch(Kb + (size_t)(j + 64 + n) * HH, 0, 3);

        // S = Q*K^T for columns [j, j+32): two 16-col tiles, K-dim 64 = 2 chunks
        const u16* kr0 = Kb + (size_t)(j + n) * HH + hi * 16;
        const u16* kr1 = Kb + (size_t)(j + 16 + n) * HH + hi * 16;
        v16bf bA0 = load_b_op(kr0);        // h 0..31, cols j..j+15
        v16bf bA1 = load_b_op(kr0 + 32);   // h 32..63
        v16bf bB0 = load_b_op(kr1);        // cols j+16..j+31
        v16bf bB1 = load_b_op(kr1 + 32);
        v8f s0 = {}, s1 = {};
        s0 = __builtin_amdgcn_wmma_f32_16x16x32_bf16(false, aq0, false, bA0, (short)0, s0, false, false);
        s0 = __builtin_amdgcn_wmma_f32_16x16x32_bf16(false, aq1, false, bA1, (short)0, s0, false, false);
        s1 = __builtin_amdgcn_wmma_f32_16x16x32_bf16(false, aq0, false, bB0, (short)0, s1, false, false);
        s1 = __builtin_amdgcn_wmma_f32_16x16x32_bf16(false, aq1, false, bB1, (short)0, s1, false, false);

        // online softmax (base-2); row M = v + 8*hi lives across 16 lanes of a half
        for (int v = 0; v < 8; ++v) {
            int row = m0 + hi * 8 + v;
            int c0 = j + n, c1 = j + 16 + n;
            float x0 = (c0 <= row) ? s0[v] * kscale : -1e30f;
            float x1 = (c1 <= row) ? s1[v] * kscale : -1e30f;
            float rm = fmaxf(x0, x1);
            rm = fmaxf(rm, __shfl_xor(rm, 1));
            rm = fmaxf(rm, __shfl_xor(rm, 2));
            rm = fmaxf(rm, __shfl_xor(rm, 4));
            rm = fmaxf(rm, __shfl_xor(rm, 8));            // stays within 16-lane half
            float mnew  = fmaxf(mi[v], rm);
            float alpha = exp2f(mi[v] - mnew);
            mi[v] = mnew;
            float p0 = exp2f(x0 - mnew);
            float p1 = exp2f(x1 - mnew);
            li[v] = li[v] * alpha + p0 + p1;
            o0[v] *= alpha; o1[v] *= alpha; o2[v] *= alpha; o3[v] *= alpha;
            myP[(hi * 8 + v) * 32 + n]      = f2bf(p0);   // P in row-major LDS
            myP[(hi * 8 + v) * 32 + 16 + n] = f2bf(p1);
        }
        // C-layout -> A-layout transpose via LDS (same wave; DS is in-order)
        asm volatile("s_wait_dscnt 0" ::: "memory");
        const u16* pr = myP + n * 32;
        v16bf pa = load_a_op(pr + hi * 16, pr + 8 + hi * 16);

        // O += P * V ; V^T rows give contiguous B-operand K-runs
        v16bf bv0 = load_b_op(Vb + (size_t)( 0 + n) * TT + j + hi * 16);
        v16bf bv1 = load_b_op(Vb + (size_t)(16 + n) * TT + j + hi * 16);
        v16bf bv2 = load_b_op(Vb + (size_t)(32 + n) * TT + j + hi * 16);
        v16bf bv3 = load_b_op(Vb + (size_t)(48 + n) * TT + j + hi * 16);
        o0 = __builtin_amdgcn_wmma_f32_16x16x32_bf16(false, pa, false, bv0, (short)0, o0, false, false);
        o1 = __builtin_amdgcn_wmma_f32_16x16x32_bf16(false, pa, false, bv1, (short)0, o1, false, false);
        o2 = __builtin_amdgcn_wmma_f32_16x16x32_bf16(false, pa, false, bv2, (short)0, o2, false, false);
        o3 = __builtin_amdgcn_wmma_f32_16x16x32_bf16(false, pa, false, bv3, (short)0, o3, false, false);
    }

    // reduce row sums across the 16-lane half (mi already half-uniform)
    for (int v = 0; v < 8; ++v) {
        float l = li[v];
        l += __shfl_xor(l, 1);
        l += __shfl_xor(l, 2);
        l += __shfl_xor(l, 4);
        l += __shfl_xor(l, 8);
        li[v] = l;
    }

    // odd wave publishes its partial (O, m, l)
    if (phase == 1) {
        for (int v = 0; v < 8; ++v) {
            int r = hi * 8 + v;
            sO[pair][r][ 0 + n] = o0[v];
            sO[pair][r][16 + n] = o1[v];
            sO[pair][r][32 + n] = o2[v];
            sO[pair][r][48 + n] = o3[v];
            if (n == 0) {
                sML[pair][0][r] = mi[v];
                sML[pair][1][r] = li[v];
            }
        }
    }
    __syncthreads();

    // even wave merges the two softmax partials and writes the result
    if (phase == 0) {
        for (int v = 0; v < 8; ++v) {
            int r = hi * 8 + v;
            float ma = mi[v],           la = li[v];
            float mb = sML[pair][0][r], lb = sML[pair][1][r];
            float mm = fmaxf(ma, mb);
            float ea = exp2f(ma - mm);
            float eb = exp2f(mb - mm);
            float inv = 1.0f / (la * ea + lb * eb);
            int row = m0 + r;
            float* orow = out + (size_t)(b * TT + row) * HH;
            orow[ 0 + n] = (o0[v] * ea + sO[pair][r][ 0 + n] * eb) * inv;
            orow[16 + n] = (o1[v] * ea + sO[pair][r][16 + n] * eb) * inv;
            orow[32 + n] = (o2[v] * ea + sO[pair][r][32 + n] * eb) * inv;
            orow[48 + n] = (o3[v] * ea + sO[pair][r][48 + n] * eb) * inv;
        }
    }
}

extern "C" void kernel_launch(void* const* d_in, const int* in_sizes, int n_in,
                              void* d_out, int out_size, void* d_ws, size_t ws_size,
                              hipStream_t stream) {
    // setup_inputs() dict order: x, Wk, Wq, Wv
    const float* x  = (const float*)d_in[0];
    const float* Wk = (const float*)d_in[1];
    const float* Wq = (const float*)d_in[2];
    const float* Wv = (const float*)d_in[3];
    float* out = (float*)d_out;

    u16* Wqt = (u16*)d_ws;                 // [H][C] bf16
    u16* Wkt = Wqt + HH * CC;
    u16* Wvt = Wkt + HH * CC;
    u16* Qbf = Wvt + HH * CC;              // [B][T][H] bf16
    u16* Kbf = Qbf + (size_t)BB * TT * HH; // [B][T][H] bf16
    u16* Vt  = Kbf + (size_t)BB * TT * HH; // [B][H][T] bf16 (transposed)

    wcvt_kernel<<<(HH * CC + 255) / 256, 256, 0, stream>>>(Wk, Wq, Wv, Wqt, Wkt, Wvt);
    qkv_proj_kernel<<<(BB * 128 * 4) / 8, 256, 0, stream>>>(x, Wqt, Wkt, Wvt, Qbf, Kbf, Vt);
    attn_kernel<<<(BB * 128) / 4, 256, 0, stream>>>(Qbf, Kbf, Vt, out);
}
